// GruopMorph_49658411876710
// MI455X (gfx1250) — matrix operand order repaired and checked by reading, not compiled, not run on hardware
//
#include <hip/hip_runtime.h>
#include <hip/hip_bf16.h>

// ---------------------------------------------------------------------------
// GroupMorph decoder on gfx1250 (MI455X).
// All convs: implicit GEMM on v_wmma_f32_16x16x32_f16, wave32.
// Activations: channels-last f16 [B, SS, 64] -> B-fragment = 2x b128 loads.
// Weights: pre-repacked A fragments; block-cooperatively staged in LDS
// (double-buffered) so all 8 waves of a block share them via ds_load_b128.
// ---------------------------------------------------------------------------

typedef __attribute__((ext_vector_type(16))) _Float16 v16h;
typedef __attribute__((ext_vector_type(8)))  float    v8f;

#define BB   8
#define DD   20
#define HH   24
#define WW   24
#define SS   (DD * HH * WW)     // 11520
#define NVOX (SS / 16)          // 720

union V32 { float4 f4[2]; v16h h; };

struct CP {
    const _Float16* xin;     // channels-last input [B, SS, 64]
    const _Float16* wfrag;   // pre-repacked A fragments
    const float*    bias;    // [Cout] or null
    _Float16*       outH;    // channels-last f16 output, or null
    float*          outF;    // f32 channels-first accumulate output (flow), or null
    int cinOff;              // channel offset into xin (16*g for splitters)
    int outStride;           // channel stride of outH
    int outOff;              // channel offset into outH
    int CoutReal;            // real out-channel count (for outF guard)
};

// ---------------------------------------------------------------------------
// Conv kernel. K flattened as q = tap*CINP + cin.
// A frag (lane,h): M=lane&15, K = kc*32 + (lane>=16?16:0) + h
// B frag (lane,h): N=lane&15, K = kc*32 + (lane>=16?16:0) + h
// D tile: lane l, vgpr r -> (m = r + 8*(l>>4), n = l&15)
// ---------------------------------------------------------------------------
template <int KSZ, int CINP, int NOC>
__global__ __launch_bounds__(256) void conv_wmma(CP p) {
    constexpr int K2  = KSZ * KSZ;
    constexpr int K3  = K2 * KSZ;
    constexpr int RAD = KSZ / 2;

    constexpr int LDSN = (CINP == 64) ? NOC * 1024 : 8;   // halves per buffer
    __shared__ _Float16 ldsA[2][LDSN];

    const int lane = threadIdx.x & 31;
    const int wav  = threadIdx.x >> 5;
    const int wid  = blockIdx.x * 8 + wav;
    if (wid >= BB * NVOX) return;            // whole wave exits together

    const int b       = wid / NVOX;
    const int voxBase = (wid % NVOX) * 16;
    const int hi      = lane >> 4;
    const int nrow    = lane & 15;
    const int vox     = voxBase + nrow;
    const int d0      = vox / (HH * WW);
    const int rr0     = vox - d0 * (HH * WW);
    const int y0      = rr0 / WW;
    const int x0      = rr0 - y0 * WW;

    v8f acc[NOC];
    #pragma unroll
    for (int t = 0; t < NOC; ++t)
        acc[t] = (v8f){0.f, 0.f, 0.f, 0.f, 0.f, 0.f, 0.f, 0.f};

    const _Float16* xb = p.xin + (long)b * SS * 64 + p.cinOff;

    if constexpr (CINP == 64) {
        constexpr int NK   = K3 * 2;
        constexpr int TOT4 = NOC * 128;                 // float4 units per tap
        constexpr int NSTG = (TOT4 + 255) / 256;        // per-thread stage count

        float4 rs[NSTG];

        // stage tap's A fragments (global -> regs)
        auto stageRegs = [&](int tap) {
            #pragma unroll
            for (int i = 0; i < NSTG; ++i) {
                const int e = (int)threadIdx.x + i * 256;
                if ((TOT4 % 256 == 0) || (e < TOT4)) {
                    const int half8  = e & 1;
                    const int lane_s = (e >> 1) & 31;
                    const int tcc    = e >> 6;          // 0 .. 2*NOC-1
                    const int t      = tcc >> 1;
                    const int cc     = tcc & 1;
                    const float4* gp = (const float4*)(
                        p.wfrag + (((long)t * NK + tap * 2 + cc) * 32 + lane_s) * 16 +
                        half8 * 8);
                    rs[i] = *gp;
                }
            }
        };
        // regs -> LDS
        auto commitRegs = [&](int buf) {
            #pragma unroll
            for (int i = 0; i < NSTG; ++i) {
                const int e = (int)threadIdx.x + i * 256;
                if ((TOT4 % 256 == 0) || (e < TOT4)) {
                    float4* lp = (float4*)(&ldsA[buf][e * 8]);
                    *lp = rs[i];
                }
            }
        };

        stageRegs(0);
        commitRegs(0);

        for (int tap = 0; tap < K3; ++tap) {
            const int cur = tap & 1;
            if (tap + 1 < K3) stageRegs(tap + 1);   // overlap latency with compute

            const int tz = tap / K2;
            const int tr = tap - tz * K2;
            const int ty = tr / KSZ;
            const int tx = tr - ty * KSZ;
            const int zd = d0 + tz - RAD, zy = y0 + ty - RAD, zx = x0 + tx - RAD;
            const bool inb = ((unsigned)zd < (unsigned)DD) &
                             ((unsigned)zy < (unsigned)HH) &
                             ((unsigned)zx < (unsigned)WW);
            const int sp = inb ? (zd * (HH * WW) + zy * WW + zx) : 0;
            const _Float16* xv = xb + (long)sp * 64 + hi * 16;

            __syncthreads();   // ldsA[cur] fully written; prev reads of other buf done

            // prefetch next tap's activation line into near caches
            if (tap + 1 < K3) {
                const int tap1 = tap + 1;
                const int tz1 = tap1 / K2;
                const int tr1 = tap1 - tz1 * K2;
                const int ty1 = tr1 / KSZ;
                const int tx1 = tr1 - ty1 * KSZ;
                const int zd1 = d0 + tz1 - RAD, zy1 = y0 + ty1 - RAD, zx1 = x0 + tx1 - RAD;
                const bool inb1 = ((unsigned)zd1 < (unsigned)DD) &
                                  ((unsigned)zy1 < (unsigned)HH) &
                                  ((unsigned)zx1 < (unsigned)WW);
                const int sp1 = inb1 ? (zd1 * (HH * WW) + zy1 * WW + zx1) : 0;
                __builtin_prefetch(xb + (long)sp1 * 64 + hi * 16, 0, 3);
            }

            #pragma unroll
            for (int cc = 0; cc < 2; ++cc) {
                V32 bv;
                const float4* bp = (const float4*)(xv + cc * 32);
                bv.f4[0] = bp[0];
                bv.f4[1] = bp[1];
                if (!inb) {
                    bv.f4[0] = make_float4(0.f, 0.f, 0.f, 0.f);
                    bv.f4[1] = make_float4(0.f, 0.f, 0.f, 0.f);
                }
                #pragma unroll
                for (int t = 0; t < NOC; ++t) {
                    V32 av;
                    const float4* ap =
                        (const float4*)(&ldsA[cur][(((t * 2 + cc) * 32) + lane) * 16]);
                    av.f4[0] = ap[0];
                    av.f4[1] = ap[1];
                    acc[t] = __builtin_amdgcn_wmma_f32_16x16x32_f16(
                                 false, av.h, false, bv.h, (short)0, acc[t], false, false);
                }
            }

            if (tap + 1 < K3) commitRegs(1 - cur);  // safe: nobody reads other buf now
        }
    } else {  // CINP == 16 (splitter convs); chunk = 2 taps, lane's tap = 2*kc + hi
        constexpr int NK = (K3 + 1) / 2;
        for (int kc = 0; kc < NK; ++kc) {
            const int  tap = 2 * kc + hi;
            const bool tok = (tap < K3);
            const int  tt  = tok ? tap : 0;
            const int tz = tt / K2;
            const int tr = tt - tz * K2;
            const int ty = tr / KSZ;
            const int tx = tr - ty * KSZ;
            const int zd = d0 + tz - RAD, zy = y0 + ty - RAD, zx = x0 + tx - RAD;
            const bool inb = tok &
                             ((unsigned)zd < (unsigned)DD) &
                             ((unsigned)zy < (unsigned)HH) &
                             ((unsigned)zx < (unsigned)WW);
            const int sp = inb ? (zd * (HH * WW) + zy * WW + zx) : 0;
            V32 bv;
            const float4* bp = (const float4*)(xb + (long)sp * 64);
            bv.f4[0] = bp[0];
            bv.f4[1] = bp[1];
            if (!inb) {
                bv.f4[0] = make_float4(0.f, 0.f, 0.f, 0.f);
                bv.f4[1] = make_float4(0.f, 0.f, 0.f, 0.f);
            }
            V32 av;
            const float4* ap = (const float4*)(p.wfrag + ((long)kc * 32 + lane) * 16);
            av.f4[0] = ap[0];
            av.f4[1] = ap[1];
            acc[0] = __builtin_amdgcn_wmma_f32_16x16x32_f16(
                         false, av.h, false, bv.h, (short)0, acc[0], false, false);
        }
    }

    // ---------------- store ----------------
    if (p.outH) {  // f16 channels-last: 8 contiguous halves per tile per lane
        #pragma unroll
        for (int t = 0; t < NOC; ++t) {
            union { _Float16 h[8]; float4 f; } s;
            #pragma unroll
            for (int r = 0; r < 8; ++r) {
                float v = acc[t][r];
                if (p.bias) v += p.bias[t * 16 + 8 * hi + r];
                s.h[r] = (_Float16)v;
            }
            _Float16* dst = p.outH + ((long)b * SS + vox) * p.outStride +
                            p.outOff + t * 16 + 8 * hi;
            *(float4*)dst = s.f;
        }
    }
    if (p.outF) {  // f32 channels-first accumulate (flow head, Cout=3)
        #pragma unroll
        for (int r = 0; r < 8; ++r) {
            const int oc = r + 8 * hi;
            if (oc < p.CoutReal) {
                float v = acc[0][r] + (p.bias ? p.bias[oc] : 0.f);
                p.outF[((long)b * p.CoutReal + oc) * SS + vox] += v;
            }
        }
    }
}

// ---------------------------------------------------------------------------
// Weight repack: w[oc][cin][tap] f32 -> A-fragment halves, zero-padded.
// dst index = ((ot*nK + kc)*32 + lane)*16 + h
// ---------------------------------------------------------------------------
__global__ __launch_bounds__(256) void repack_weights(
        const float* __restrict__ w, _Float16* __restrict__ dst,
        int Cout, int CinReal, int CINP, int k3, int nOc, int nK) {
    long idx = (long)blockIdx.x * 256 + threadIdx.x;
    const long total = (long)nOc * nK * 512;
    if (idx >= total) return;
    const int h    = (int)(idx & 15);
    const int lane = (int)((idx >> 4) & 31);
    const long g   = idx >> 9;
    const int kc   = (int)(g % nK);
    const int ot   = (int)(g / nK);
    const int m    = lane & 15;
    const int q    = kc * 32 + ((lane >= 16) ? 16 : 0) + h;
    const int oc   = ot * 16 + m;
    const int tap  = q / CINP;
    const int cin  = q - tap * CINP;
    float v = 0.f;
    if (oc < Cout && tap < k3 && cin < CinReal)
        v = w[((long)oc * CinReal + cin) * k3 + tap];
    dst[idx] = (_Float16)v;
}

// f32 channels-first -> f16 channels-last [B, SS, 64]
__global__ __launch_bounds__(256) void to_cl_f16(
        const float* __restrict__ in, _Float16* __restrict__ out) {
    long idx = (long)blockIdx.x * 256 + threadIdx.x;
    const long total = (long)BB * SS * 64;
    if (idx >= total) return;
    const int c = (int)(idx & 63);
    const long bv = idx >> 6;
    const int v = (int)(bv % SS);
    const int b = (int)(bv / SS);
    out[idx] = (_Float16)in[((long)b * 64 + c) * SS + v];
}

// correlation volume, channels-last: out ch 16..42 of xbuf
__global__ __launch_bounds__(256) void corr_cl(
        const _Float16* xm /* xbuf ch0..15 */, const _Float16* ff /* [B,SS,16] */,
        _Float16* outx /* xbuf ch16..42 */) {
    const int idx = blockIdx.x * 256 + threadIdx.x;
    if (idx >= BB * SS) return;
    const int b = idx / SS, vox = idx % SS;
    const int d = vox / (HH * WW);
    const int r = vox - d * (HH * WW);
    const int y = r / WW;
    const int x = r - y * WW;
    V32 mv;
    const float4* mp = (const float4*)(xm + ((long)b * SS + vox) * 64);
    mv.f4[0] = mp[0];
    mv.f4[1] = mp[1];
    float mf[16];
    #pragma unroll
    for (int c = 0; c < 16; ++c) mf[c] = (float)mv.h[c];
    _Float16* dst = outx + ((long)b * SS + vox) * 64 + 16;
    int s = 0;
    #pragma unroll
    for (int i = 0; i < 3; ++i)
        for (int j = 0; j < 3; ++j)
            for (int kk = 0; kk < 3; ++kk, ++s) {
                const int zd = d + i - 1, zy = y + j - 1, zx = x + kk - 1;
                float a = 0.f;
                if (((unsigned)zd < (unsigned)DD) & ((unsigned)zy < (unsigned)HH) &
                    ((unsigned)zx < (unsigned)WW)) {
                    V32 fv;
                    const float4* fp = (const float4*)(
                        ff + ((long)b * SS + zd * (HH * WW) + zy * WW + zx) * 16);
                    fv.f4[0] = fp[0];
                    fv.f4[1] = fp[1];
                    #pragma unroll
                    for (int c = 0; c < 16; ++c) a += mf[c] * (float)fv.h[c];
                    a *= (1.f / 16.f);
                }
                dst[s] = (_Float16)a;
            }
}

// InstanceNorm stats over channels-last f16; one block per (b,c)
__global__ __launch_bounds__(256) void inorm_stats_cl(
        const _Float16* __restrict__ x, int Cp, float* __restrict__ stats) {
    const int bc = blockIdx.x;
    const int b = bc / Cp, c = bc % Cp;
    const _Float16* p = x + (long)b * SS * Cp + c;
    float s = 0.f, s2 = 0.f;
    for (int v = threadIdx.x; v < SS; v += 256) {
        const float t = (float)p[(long)v * Cp];
        s += t; s2 += t * t;
    }
    __shared__ float sh[256], sh2[256];
    sh[threadIdx.x] = s; sh2[threadIdx.x] = s2;
    __syncthreads();
    for (int o = 128; o > 0; o >>= 1) {
        if (threadIdx.x < o) {
            sh[threadIdx.x]  += sh[threadIdx.x + o];
            sh2[threadIdx.x] += sh2[threadIdx.x + o];
        }
        __syncthreads();
    }
    if (threadIdx.x == 0) {
        const float mean = sh[0] / (float)SS;
        float var = sh2[0] / (float)SS - mean * mean;
        if (var < 0.f) var = 0.f;
        stats[2 * bc]     = mean;
        stats[2 * bc + 1] = rsqrtf(var + 1e-5f);
    }
}

// normalized + LeakyReLU copy (64-channel buffers), f16 cl -> f16 cl
__global__ __launch_bounds__(256) void norm_lrelu_cl(
        const _Float16* __restrict__ x, const float* __restrict__ stats,
        _Float16* __restrict__ y) {
    long idx = (long)blockIdx.x * 256 + threadIdx.x;
    const long total = (long)BB * SS * 64;
    if (idx >= total) return;
    const int c = (int)(idx & 63);
    const int b = (int)((idx >> 6) / SS);
    float v = (float)x[idx];
    const float* st = stats + 2 * ((long)b * 64 + c);
    v = (v - st[0]) * st[1];
    v = (v < 0.f) ? 0.2f * v : v;
    y[idx] = (_Float16)v;
}

// context head: IN + LReLU, f16 cl [B,SS,32] -> f32 d_out channels-first
__global__ __launch_bounds__(256) void ctx_write(
        const _Float16* __restrict__ x, const float* __restrict__ stats,
        float* __restrict__ out, int chOff) {
    long idx = (long)blockIdx.x * 256 + threadIdx.x;
    const long total = (long)BB * 32 * SS;
    if (idx >= total) return;
    const int v = (int)(idx % SS);
    const long t = idx / SS;
    const int c = (int)(t % 32);
    const int b = (int)(t / 32);
    float val = (float)x[((long)b * SS + v) * 32 + c];
    const float* st = stats + 2 * ((long)b * 32 + c);
    val = (val - st[0]) * st[1];
    val = (val < 0.f) ? 0.2f * val : val;
    out[((long)b * 131 + chOff + c) * SS + v] = val;
}

__global__ __launch_bounds__(256) void zero_f32(float* __restrict__ p, long n) {
    long idx = (long)blockIdx.x * 256 + threadIdx.x;
    if (idx < n) p[idx] = 0.f;
}

// zero xbuf pad channels 43..63
__global__ __launch_bounds__(256) void zero_pad_cl(_Float16* __restrict__ x) {
    long idx = (long)blockIdx.x * 256 + threadIdx.x;
    const long total = (long)BB * SS * 21;
    if (idx >= total) return;
    const int cp = (int)(idx % 21);
    const long bv = idx / 21;
    x[bv * 64 + 43 + cp] = (_Float16)0.f;
}

__global__ __launch_bounds__(256) void flow_avg_kernel(
        const float* __restrict__ accum, float* __restrict__ out) {
    long idx = (long)blockIdx.x * 256 + threadIdx.x;
    const long total = (long)BB * 3 * SS;
    if (idx >= total) return;
    const int v = (int)(idx % SS);
    const long t = idx / SS;
    const int c = (int)(t % 3);
    const int b = (int)(t / 3);
    out[((long)b * 131 + c) * SS + v] = accum[idx] * 0.25f;
}

// ---------------------------------------------------------------------------
// Host side
// ---------------------------------------------------------------------------
template <int KSZ, int CINP, int NOC>
static void launch_conv(hipStream_t s, const _Float16* xin, int cinOff,
                        const _Float16* wfrag, const float* bias,
                        _Float16* outH, int outStride, int outOff,
                        float* outF, int CoutReal) {
    CP p;
    p.xin = xin; p.wfrag = wfrag; p.bias = bias;
    p.outH = outH; p.outF = outF;
    p.cinOff = cinOff; p.outStride = outStride; p.outOff = outOff;
    p.CoutReal = CoutReal;
    conv_wmma<KSZ, CINP, NOC><<<(BB * NVOX) / 8, 256, 0, s>>>(p);
}

static void launch_repack(hipStream_t s, const float* w, _Float16* dst,
                          int Cout, int CinReal, int CINP, int k3, int nOc, int nK) {
    const long n = (long)nOc * nK * 512;
    repack_weights<<<(int)((n + 255) / 256), 256, 0, s>>>(
        w, dst, Cout, CinReal, CINP, k3, nOc, nK);
}

extern "C" void kernel_launch(void* const* d_in, const int* in_sizes, int n_in,
                              void* d_out, int out_size, void* d_ws, size_t ws_size,
                              hipStream_t stream) {
    (void)in_sizes; (void)n_in; (void)out_size; (void)ws_size;

    const float* m8 = (const float*)d_in[0];
    const float* f8 = (const float*)d_in[1];
    const float* sw[4] = {(const float*)d_in[2], (const float*)d_in[4],
                          (const float*)d_in[6], (const float*)d_in[8]};
    const float* sb[4] = {(const float*)d_in[3], (const float*)d_in[5],
                          (const float*)d_in[7], (const float*)d_in[9]};
    const float* w1 = (const float*)d_in[10]; const float* b1 = (const float*)d_in[11];
    const float* w2 = (const float*)d_in[12]; const float* b2 = (const float*)d_in[13];
    const float* wlist[4] = {(const float*)d_in[14], (const float*)d_in[16],
                             (const float*)d_in[18], (const float*)d_in[20]};
    const float* blist[4] = {(const float*)d_in[15], (const float*)d_in[17],
                             (const float*)d_in[19], (const float*)d_in[21]};
    const float* wf = (const float*)d_in[22]; const float* bf = (const float*)d_in[23];
    const float* wc = (const float*)d_in[24];
    float* out = (float*)d_out;

    // ---- workspace layout (bytes) ----
    char* base = (char*)d_ws;
    size_t off = 0;
    auto take = [&](size_t bytes) {
        void* p = base + off;
        off += (bytes + 255) & ~(size_t)255;
        return p;
    };
    const size_t cl64 = (size_t)BB * SS * 64 * sizeof(_Float16);
    _Float16* mh      = (_Float16*)take(cl64);
    _Float16* fh      = (_Float16*)take(cl64);
    _Float16* xbuf    = (_Float16*)take(cl64);                       // [B,SS,64]
    _Float16* fbuf    = (_Float16*)take((size_t)BB * SS * 16 * 2);   // [B,SS,16]
    _Float16* bufA    = (_Float16*)take(cl64);
    _Float16* bufB    = (_Float16*)take(cl64);
    _Float16* xn      = (_Float16*)take(cl64);
    _Float16* ctmp    = (_Float16*)take((size_t)BB * SS * 32 * 2);   // [B,SS,32]
    float*    flowacc = (float*)take((size_t)BB * 3 * SS * 4);
    float*    statsb  = (float*)take((size_t)BB * 64 * 2 * 4);

    // weight fragment buffers
    const int nK64 = 54;  // 27 taps * 2 chunks
    const int k3s[4]  = {1, 27, 125, 343};
    const int nKs[4]  = {1, 14, 63, 172};
    _Float16* swf[4];
    for (int g = 0; g < 4; ++g) swf[g] = (_Float16*)take((size_t)nKs[g] * 512 * 2);
    _Float16* w1f = (_Float16*)take((size_t)4 * nK64 * 512 * 2);
    _Float16* wlf[4];
    for (int j = 0; j < 4; ++j) wlf[j] = (_Float16*)take((size_t)4 * nK64 * 512 * 2);
    _Float16* w2f = (_Float16*)take((size_t)4 * nK64 * 512 * 2);
    _Float16* wff = (_Float16*)take((size_t)1 * nK64 * 512 * 2);
    _Float16* wcf = (_Float16*)take((size_t)2 * nK64 * 512 * 2);

    // ---- preprocessing ----
    {
        const long n = (long)BB * SS * 64;
        to_cl_f16<<<(int)((n + 255) / 256), 256, 0, stream>>>(m8, mh);
        to_cl_f16<<<(int)((n + 255) / 256), 256, 0, stream>>>(f8, fh);
        const long np = (long)BB * SS * 21;
        zero_pad_cl<<<(int)((np + 255) / 256), 256, 0, stream>>>(xbuf);
        const long nz = (long)BB * 3 * SS;
        zero_f32<<<(int)((nz + 255) / 256), 256, 0, stream>>>(flowacc, nz);
    }
    for (int g = 0; g < 4; ++g)
        launch_repack(stream, sw[g], swf[g], 16, 16, 16, k3s[g], 1, nKs[g]);
    launch_repack(stream, w1, w1f, 64, 43, 64, 27, 4, nK64);
    launch_repack(stream, w2, w2f, 64, 64, 64, 27, 4, nK64);
    for (int j = 0; j < 4; ++j)
        launch_repack(stream, wlist[j], wlf[j], 64, 64, 64, 27, 4, nK64);
    launch_repack(stream, wf, wff, 3, 64, 64, 27, 1, nK64);
    launch_repack(stream, wc, wcf, 32, 64, 64, 27, 2, nK64);

    // ---- per-group pipeline ----
    for (int g = 0; g < 4; ++g) {
        // splitter convs
        switch (g) {
        case 0:
            launch_conv<1, 16, 1>(stream, mh, 16 * g, swf[g], sb[g], xbuf, 64, 0, nullptr, 16);
            launch_conv<1, 16, 1>(stream, fh, 16 * g, swf[g], sb[g], fbuf, 16, 0, nullptr, 16);
            break;
        case 1:
            launch_conv<3, 16, 1>(stream, mh, 16 * g, swf[g], sb[g], xbuf, 64, 0, nullptr, 16);
            launch_conv<3, 16, 1>(stream, fh, 16 * g, swf[g], sb[g], fbuf, 16, 0, nullptr, 16);
            break;
        case 2:
            launch_conv<5, 16, 1>(stream, mh, 16 * g, swf[g], sb[g], xbuf, 64, 0, nullptr, 16);
            launch_conv<5, 16, 1>(stream, fh, 16 * g, swf[g], sb[g], fbuf, 16, 0, nullptr, 16);
            break;
        default:
            launch_conv<7, 16, 1>(stream, mh, 16 * g, swf[g], sb[g], xbuf, 64, 0, nullptr, 16);
            launch_conv<7, 16, 1>(stream, fh, 16 * g, swf[g], sb[g], fbuf, 16, 0, nullptr, 16);
            break;
        }

        // correlation volume -> xbuf ch16..42
        {
            const int n = BB * SS;
            corr_cl<<<(n + 255) / 256, 256, 0, stream>>>(xbuf, fbuf, xbuf);
        }

        // conv1 (43->64 via 64-padded channels) -> bufA
        launch_conv<3, 64, 4>(stream, xbuf, 0, w1f, b1, bufA, 64, 0, nullptr, 64);
        // IN + LReLU -> xn ; conv w2 -> bufB
        inorm_stats_cl<<<BB * 64, 256, 0, stream>>>(bufA, 64, statsb);
        {
            const long n = (long)BB * SS * 64;
            norm_lrelu_cl<<<(int)((n + 255) / 256), 256, 0, stream>>>(bufA, statsb, xn);
        }
        launch_conv<3, 64, 4>(stream, xn, 0, w2f, b2, bufB, 64, 0, nullptr, 64);

        _Float16* cur = bufB;
        _Float16* oth = bufA;
        const int extra = (g == 0) ? 0 : (g == 1) ? 2 : (g == 2) ? 3 : 4;
        for (int j = 0; j < extra; ++j) {
            inorm_stats_cl<<<BB * 64, 256, 0, stream>>>(cur, 64, statsb);
            const long n = (long)BB * SS * 64;
            norm_lrelu_cl<<<(int)((n + 255) / 256), 256, 0, stream>>>(cur, statsb, xn);
            launch_conv<3, 64, 4>(stream, xn, 0, wlf[j], blist[j], oth, 64, 0, nullptr, 64);
            _Float16* t = cur; cur = oth; oth = t;
        }

        // flow head -> accumulate f32
        launch_conv<3, 64, 1>(stream, cur, 0, wff, bf, nullptr, 0, 0, flowacc, 3);

        // context head -> ctmp, then IN+LReLU -> d_out
        launch_conv<3, 64, 2>(stream, cur, 0, wcf, nullptr, ctmp, 32, 0, nullptr, 32);
        inorm_stats_cl<<<BB * 32, 256, 0, stream>>>(ctmp, 32, statsb);
        {
            const long n = (long)BB * 32 * SS;
            ctx_write<<<(int)((n + 255) / 256), 256, 0, stream>>>(
                ctmp, statsb, out, 3 + 32 * g);
        }
    }

    // flow = mean of 4 group flows
    {
        const long n = (long)BB * 3 * SS;
        flow_avg_kernel<<<(int)((n + 255) / 256), 256, 0, stream>>>(flowacc, out);
    }
}
